// macroDiscriminator_80633716015376
// MI455X (gfx1250) — compile-verified, matrix operand order checked by reading, not run on hardware
//
#include <hip/hip_runtime.h>

// ---- problem constants ----
#define N_H     89
#define N_3H    267
#define T_STEPS 120
#define B_TOT   32768

// ---- tiling ----
#define HP  96      // K padded to 3*32
#define NP  272     // 3H padded to 17*16
#define WS  104     // LDS row stride (bf16) for X/H/W  (16B aligned, bank-conflict free)
#define GS  276     // LDS row stride (f32) for gate buffers
#define MT  64      // batch rows per block
#define NWAVES 8
#define NTILES 68   // (MT/16) * (NP/16)

typedef __attribute__((ext_vector_type(16))) __bf16 v16bf;
typedef __attribute__((ext_vector_type(8)))  float  v8f;

__device__ __forceinline__ __bf16 f2bf(float f) {
  unsigned u = __builtin_bit_cast(unsigned, f);
  u += 0x7fffu + ((u >> 16) & 1u);               // round-to-nearest-even
  return __builtin_bit_cast(__bf16, (unsigned short)(u >> 16));
}

__device__ __forceinline__ float sigm(float x) { return 1.0f / (1.0f + __expf(-x)); }

// 16x32 bf16 fragment: lane L holds row/col (L&15), K = kbase..+7 and kbase+16..+23
__device__ __forceinline__ v16bf load_frag(const __bf16* p) {
  v16bf v;
  ((uint4*)&v)[0] = *(const uint4*)(p);
  ((uint4*)&v)[1] = *(const uint4*)(p + 16);
  return v;
}

// fp32 weights [3H, H] -> bf16 padded [NP, HP], exact zeros in padded K/N
__global__ void prep_weights(const float* __restrict__ Wih,
                             const float* __restrict__ Whh,
                             __bf16* __restrict__ wbf) {
  int idx = blockIdx.x * blockDim.x + threadIdx.x;
  const int per = NP * HP;
  if (idx >= 2 * per) return;
  int which = idx / per;
  int rem   = idx - which * per;
  int n = rem / HP, k = rem - n * HP;
  const float* W = which ? Whh : Wih;
  float v = (n < N_3H && k < N_H) ? W[n * N_H + k] : 0.0f;
  wbf[idx] = f2bf(v);
}

__global__ __launch_bounds__(256)
void gru_kernel(const float* __restrict__ X,      // (B, T, H)
                const float* __restrict__ b_ih,
                const float* __restrict__ b_hh,
                const float* __restrict__ W_m,    // (32, H)
                const float* __restrict__ b_m,
                const float* __restrict__ W_o,    // (1, 32)
                const float* __restrict__ b_o,
                const __bf16* __restrict__ wbf,   // [2][NP][HP] bf16
                float* __restrict__ outp) {       // [B] scores ++ [B*32] metric
  __shared__ __align__(16) __bf16 sWih[NP * WS];
  __shared__ __align__(16) __bf16 sWhh[NP * WS];
  __shared__ __align__(16) __bf16 sXbuf[2][MT * WS];   // double-buffered x tile
  __shared__ __align__(16) __bf16 sH[MT * WS];
  __shared__ float sHf[MT * HP];
  __shared__ float sGi[MT * GS];
  __shared__ float sGh[MT * GS];
  __shared__ float sBi[NP];
  __shared__ float sBh[NP];

  const int tid  = threadIdx.x;
  const int lane = tid & 31;                 // wave32
  const int wave = tid >> 5;
  const int b0   = blockIdx.x * MT;

  // ---- one-time staging ----
  for (int i = tid; i < NP * HP; i += 256) {
    int n = i / HP, k = i - n * HP;
    sWih[n * WS + k] = wbf[i];
    sWhh[n * WS + k] = wbf[NP * HP + i];
  }
  for (int i = tid; i < MT * WS; i += 256) sH[i] = f2bf(0.0f);
  for (int i = tid; i < MT * HP; i += 256) sHf[i] = 0.0f;
  for (int i = tid; i < NP; i += 256) {
    sBi[i] = (i < N_3H) ? b_ih[i] : 0.0f;
    sBh[i] = (i < N_3H) ? b_hh[i] : 0.0f;
  }
  // preload x_0 into buffer 0
  for (int i = tid; i < MT * HP; i += 256) {
    int m = i / HP, k = i - m * HP;
    float v = (k < N_H) ? X[(size_t)(b0 + m) * T_STEPS * N_H + k] : 0.0f;
    sXbuf[0][m * WS + k] = f2bf(v);
  }
  __syncthreads();

  const int mfrag = lane & 15;               // row (A) / col (B) within tile
  const int khalf = (lane >> 4) << 3;        // K phase: 0 or 8
  const int crow  = (lane >> 4) << 3;        // C/D: upper lanes hold rows +8
  const int mt    = wave & 3;                // M-tile is wave-invariant (stride 8 over 68)
  const int pm    = tid >> 2;                // prefetch: row per thread group
  const int pc    = tid & 3;                 // prefetch: 128B chunk within row

  for (int t = 0; t < T_STEPS; ++t) {
    const __bf16* cur = sXbuf[t & 1];
    __bf16*       nxt = sXbuf[(t & 1) ^ 1];

    // prefetch x_{t+1}: 64 rows x <=4 cachelines; speculative, no VGPR payload,
    // the WMMA work below covers the HBM->L2/WGP$ latency
    if (t + 1 < T_STEPS) {
      const float* p = &X[((size_t)(b0 + pm) * T_STEPS + (t + 1)) * N_H + pc * 32];
      __builtin_prefetch(p, 0, 3);
    }

    // hoisted A fragments for this wave's M-tile
    const __bf16* arow = &cur[(mt * 16 + mfrag) * WS];
    const __bf16* hrow = &sH [(mt * 16 + mfrag) * WS];
    v16bf axf[3], ahf[3];
#pragma unroll
    for (int kt = 0; kt < 3; ++kt) {
      axf[kt] = load_frag(arow + kt * 32 + khalf);
      ahf[kt] = load_frag(hrow + kt * 32 + khalf);
    }

    // gi = x @ Wih^T, gh = h @ Whh^T  (bf16 WMMA, f32 accum)
    for (int tl = wave; tl < NTILES; tl += NWAVES) {
      int nt = tl >> 2;
      const __bf16* birow = &sWih[(nt * 16 + mfrag) * WS];
      const __bf16* bhrow = &sWhh[(nt * 16 + mfrag) * WS];
      v8f accI = {};
      v8f accH = {};
#pragma unroll
      for (int kt = 0; kt < 3; ++kt) {
        int k0 = kt * 32 + khalf;
        v16bf bi = load_frag(birow + k0);
        accI = __builtin_amdgcn_wmma_f32_16x16x32_bf16(false, axf[kt], false, bi,
                                                       (short)0, accI, false, false);
        v16bf bh = load_frag(bhrow + k0);
        accH = __builtin_amdgcn_wmma_f32_16x16x32_bf16(false, ahf[kt], false, bh,
                                                       (short)0, accH, false, false);
      }
      int n     = nt * 16 + mfrag;
      int mbase = mt * 16 + crow;
#pragma unroll
      for (int r = 0; r < 8; ++r) {
        sGi[(mbase + r) * GS + n] = accI[r];
        sGh[(mbase + r) * GS + n] = accH[r];
      }
    }

    // stage x_{t+1} (cache-hot after prefetch) into the alternate buffer;
    // no reader until two barriers later
    if (t + 1 < T_STEPS) {
      for (int i = tid; i < MT * HP; i += 256) {
        int m = i / HP, k = i - m * HP;
        float v = (k < N_H)
                    ? X[((size_t)(b0 + m) * T_STEPS + (t + 1)) * N_H + k] : 0.0f;
        nxt[m * WS + k] = f2bf(v);
      }
    }
    __syncthreads();

    // gate elementwise + h update (PyTorch GRUCell semantics)
    for (int i = tid; i < MT * N_H; i += 256) {
      int m = i / N_H, n = i - m * N_H;
      float ir  = sGi[m * GS + n]            + sBi[n];
      float hr  = sGh[m * GS + n]            + sBh[n];
      float iz  = sGi[m * GS + n + N_H]      + sBi[n + N_H];
      float hz  = sGh[m * GS + n + N_H]      + sBh[n + N_H];
      float in_ = sGi[m * GS + n + 2 * N_H]  + sBi[n + 2 * N_H];
      float hn  = sGh[m * GS + n + 2 * N_H]  + sBh[n + 2 * N_H];
      float r  = sigm(ir + hr);
      float z  = sigm(iz + hz);
      float nn = tanhf(in_ + r * hn);
      float hp = sHf[m * HP + n];
      float hv = (1.0f - z) * nn + z * hp;
      sHf[m * HP + n] = hv;
      sH [m * WS + n] = f2bf(hv);
    }
    __syncthreads();
  }

  // metric_fc: sigmoid(h @ W_m^T + b_m) -> (MT, 32); stash in sGi (reused)
  for (int i = tid; i < MT * 32; i += 256) {
    int m = i >> 5, c = i & 31;
    float acc = b_m[c];
    for (int k = 0; k < N_H; ++k) acc += sHf[m * HP + k] * W_m[c * N_H + k];
    float v = sigm(acc);
    outp[(size_t)B_TOT + (size_t)(b0 + m) * 32 + c] = v;
    sGi[m * 32 + c] = v;
  }
  __syncthreads();

  // output_fc: sigmoid(metric @ W_o^T + b_o) -> (MT, 1)
  for (int m = tid; m < MT; m += 256) {
    float acc = b_o[0];
    for (int c = 0; c < 32; ++c) acc += sGi[m * 32 + c] * W_o[c];
    outp[b0 + m] = sigm(acc);
  }
}

extern "C" void kernel_launch(void* const* d_in, const int* in_sizes, int n_in,
                              void* d_out, int out_size, void* d_ws, size_t ws_size,
                              hipStream_t stream) {
  (void)in_sizes; (void)n_in; (void)out_size; (void)ws_size;
  const float* X   = (const float*)d_in[0];
  const float* Wih = (const float*)d_in[1];
  const float* Whh = (const float*)d_in[2];
  const float* bih = (const float*)d_in[3];
  const float* bhh = (const float*)d_in[4];
  const float* Wm  = (const float*)d_in[5];
  const float* bm  = (const float*)d_in[6];
  const float* Wo  = (const float*)d_in[7];
  const float* bo  = (const float*)d_in[8];

  __bf16* wbf = (__bf16*)d_ws;               // needs 2*NP*HP*2 = 104,448 bytes
  int total = 2 * NP * HP;
  prep_weights<<<(total + 255) / 256, 256, 0, stream>>>(Wih, Whh, wbf);
  gru_kernel<<<B_TOT / MT, 256, 0, stream>>>(X, bih, bhh, Wm, bm, Wo, bo, wbf,
                                             (float*)d_out);
}
